// DGCNNPaper_47193100648615
// MI455X (gfx1250) — matrix operand order repaired and checked by reading, not compiled, not run on hardware
//
#include <hip/hip_runtime.h>
#include <hip/hip_bf16.h>
#include <math.h>

// ---------------- problem constants ----------------
#define BB 16
#define NN 1024
#define KNN 20
#define LIN 1024
#define HID 512
#define OUTD 40

// ---------------- types ----------------
typedef __attribute__((ext_vector_type(16))) __bf16         v16bf;
typedef __attribute__((ext_vector_type(8)))  float          v8f;
typedef __attribute__((ext_vector_type(8)))  unsigned short u16x8;

union FragU { v16bf v; u16x8 h[2]; };

// ---------------- small helpers ----------------
__device__ inline unsigned short f2bf(float f) {
    unsigned u = __float_as_uint(f);
    unsigned r = (u + 0x7FFFu + ((u >> 16) & 1u)) >> 16;   // RNE
    return (unsigned short)r;
}
__device__ inline float bf2f(unsigned short h) {
    return __uint_as_float(((unsigned)h) << 16);
}
// monotone float<->uint mapping for atomicMax/Min on floats
__device__ inline unsigned ford(float f) {
    unsigned u = __float_as_uint(f);
    return (u & 0x80000000u) ? ~u : (u | 0x80000000u);
}
__device__ inline float funord(unsigned u) {
    return (u & 0x80000000u) ? __uint_as_float(u ^ 0x80000000u)
                             : __uint_as_float(~u);
}

__device__ inline v8f wmma_bf16(v16bf a, v16bf b, v8f c) {
    // D = A(16x32 bf16) * B(32x16 bf16) + C(16x16 f32)
    return __builtin_amdgcn_wmma_f32_16x16x32_bf16(
        /*neg_a=*/false, a, /*neg_b=*/false, b,
        /*c_mod=*/(short)0, c, /*reuse_a=*/false, /*reuse_b=*/false);
}

// A fragment (16x32 bf16), A row-major with leading dim lda (elements).
// ISA layout: lane<16 -> M=lane, K={0..7,16..23}; lane>=16 -> M=lane-16, K={8..15,24..31}
__device__ inline v16bf load_a_frag(const unsigned short* __restrict__ A,
                                    int lda, int m0, int k0) {
    int lane = threadIdx.x & 31;
    int m    = lane & 15;
    int koff = (lane >> 4) * 8;
    const unsigned short* p = A + (size_t)(m0 + m) * lda + (k0 + koff);
    FragU f;
    f.h[0] = *(const u16x8*)(p);
    f.h[1] = *(const u16x8*)(p + 16);
    return f.v;
}

// B fragment (32x16 bf16), B row-major (K x N) leading dim ldb.
// lane<16 -> N=lane, K={0..7,16..23}; lane>=16 -> N=lane-16, K={8..15,24..31}
__device__ inline v16bf load_b_frag(const unsigned short* __restrict__ Bm,
                                    int ldb, int k0, int n0) {
    int lane = threadIdx.x & 31;
    int koff = (lane >> 4) * 8;
    const unsigned short* p = Bm + (size_t)(k0 + koff) * ldb + n0 + (lane & 15);
    FragU f;
#pragma unroll
    for (int s = 0; s < 8; ++s) {
        f.h[0][s] = p[(size_t)s * ldb];
        f.h[1][s] = p[(size_t)(16 + s) * ldb];
    }
    return f.v;
}

// CDNA5 async global->LDS copy (16B per lane), tracked by ASYNCcnt.
__device__ inline void async_copy_b128(const unsigned short* g, unsigned short* lds) {
    unsigned ldsoff = (unsigned)(uintptr_t)lds;        // LDS aperture: low 32 bits = LDS addr
    asm volatile("global_load_async_to_lds_b128 %0, %1, off"
                 :: "v"(ldsoff), "v"(g) : "memory");
}
__device__ inline void async_wait0() {
    asm volatile("s_wait_asynccnt 0x0" ::: "memory");
}

// ---------------- K0: init stats buffers ----------------
__global__ void k_init(float* colsum, float* colsumsq,
                       unsigned* ymax, unsigned* ymin) {
    int i = blockIdx.x * blockDim.x + threadIdx.x;
    if (i < LIN) { colsum[i] = 0.f; colsumsq[i] = 0.f; }
    if (i < BB * LIN) { ymax[i] = ford(-INFINITY); ymin[i] = ford(INFINITY); }
}

// ---------------- K1: fp32 -> bf16 row-major + transposed + sqnorm + concat slice ----------------
__global__ void k_cast(const float* __restrict__ x, int d, int dpad,
                       unsigned short* __restrict__ xb,
                       unsigned short* __restrict__ xbT,
                       float* __restrict__ sq,
                       unsigned short* __restrict__ cat, int catoff) {
    int i = blockIdx.x * blockDim.x + threadIdx.x;   // point index in [0, B*N)
    if (i >= BB * NN) return;
    int b = i >> 10, n = i & 1023;
    const float* xi = x + (size_t)i * d;
    float s = 0.f;
    for (int c = 0; c < dpad; ++c) {
        float v = (c < d) ? xi[c] : 0.f;
        s += v * v;
        unsigned short h = f2bf(v);
        xb[(size_t)i * dpad + c] = h;
        xbT[((size_t)b * dpad + c) * NN + n] = h;
        if (cat && c < d) cat[(size_t)i * 512 + catoff + c] = h;
    }
    sq[i] = s;
}

// ---------------- K2: weight fp32 -> bf16 with K padded to kpad ----------------
__global__ void k_wcast(const float* __restrict__ W, int kdim, int kpad, int ndim,
                        unsigned short* __restrict__ wb) {
    int i = blockIdx.x * blockDim.x + threadIdx.x;
    int total = kpad * ndim;
    if (i >= total) return;
    int kk = i / ndim, n = i - kk * ndim;
    wb[i] = f2bf(kk < kdim ? W[(size_t)kk * ndim + n] : 0.f);
}

// ---------------- K3: Gram matrix G = X * X^T per cloud (WMMA bf16) ----------------
// grid (NN/64, NN/32, B), block 256 = 8 waves arranged 4(m) x 2(n)
template <int DPAD>
__global__ void k_gram(const unsigned short* __restrict__ xb,
                       const unsigned short* __restrict__ xbT,
                       float* __restrict__ gram) {
    int b = blockIdx.z;
    int w = threadIdx.x >> 5, lane = threadIdx.x & 31;
    int m0 = blockIdx.x * 64 + (w >> 1) * 16;
    int n0 = blockIdx.y * 32 + (w & 1) * 16;
    const unsigned short* A  = xb  + (size_t)b * NN * DPAD;
    const unsigned short* Bt = xbT + (size_t)b * DPAD * NN;
    v8f acc = {};
#pragma unroll
    for (int k0 = 0; k0 < DPAD; k0 += 32) {
        __builtin_prefetch(A + (size_t)m0 * DPAD + k0 + 32, 0, 1);
        __builtin_prefetch(Bt + (size_t)(k0 + 32) * NN + n0, 0, 1);
        v16bf a  = load_a_frag(A, DPAD, m0, k0);
        v16bf bf = load_b_frag(Bt, NN, k0, n0);
        acc = wmma_bf16(a, bf, acc);
    }
    int n = n0 + (lane & 15);
    float* G = gram + (size_t)b * NN * NN;
#pragma unroll
    for (int v = 0; v < 8; ++v) {
        int m = m0 + v + ((lane >> 4) << 3);
        G[(size_t)m * NN + n] = acc[v];
    }
}

// ---------------- K4: top-k (smallest distance) via register insertion ----------------
__global__ void k_topk(const float* __restrict__ gram, const float* __restrict__ sq,
                       int* __restrict__ idx) {
    int i = blockIdx.x * blockDim.x + threadIdx.x;   // [0, B*N)
    if (i >= BB * NN) return;
    int b = i >> 10, r = i & 1023;
    const float* G   = gram + (size_t)b * NN * NN + (size_t)r * NN;
    const float* sqb = sq + b * NN;
    float si = sqb[r];
    float bd[KNN]; int bi[KNN];
#pragma unroll
    for (int t = 0; t < KNN; ++t) { bd[t] = INFINITY; bi[t] = 0; }
    for (int j = 0; j < NN; ++j) {
        float d = si + sqb[j] - 2.f * G[j];
        if (d < bd[KNN - 1]) {
            int pos = KNN - 1;
#pragma unroll
            for (int s = KNN - 1; s > 0; --s) {
                if (d < bd[s - 1]) { bd[s] = bd[s - 1]; bi[s] = bi[s - 1]; pos = s - 1; }
            }
            bd[pos] = d; bi[pos] = j;
        }
    }
#pragma unroll
    for (int t = 0; t < KNN; ++t) idx[(size_t)i * KNN + t] = bi[t];
}

// ---------------- K5: edge MLP + max over neighbors (WMMA bf16, A from LDS) ----------
// grid (NN/16, B), block 256 = 8 waves; wave w owns output n-tiles {w, w+8, ...}
template <int DIN, int DPAD, int KPAD, int DOUT>
__global__ void k_edge(const unsigned short* __restrict__ xb,
                       const int* __restrict__ idx,
                       const unsigned short* __restrict__ wb,
                       const float* __restrict__ bias,
                       float* __restrict__ xout) {
    constexpr int NTILES = DOUT / 16;
    constexpr int TPW    = (NTILES + 7) / 8;      // n-tiles per wave
    constexpr int NCH    = KPAD / 32;             // K chunks
    constexpr bool HOIST = (NCH * TPW) <= 4;      // keep B-fragments in registers?
    constexpr int BHN    = HOIST ? NCH * TPW : 1;

    __shared__ unsigned short E[16 * KPAD];       // 16 points x edge features
    __shared__ int Jn[16];                        // neighbor index per point

    int b   = blockIdx.y;
    int i0  = blockIdx.x * 16;
    int tid = threadIdx.x;
    int w = tid >> 5, lane = tid & 31;
    const unsigned short* Xb = xb + (size_t)b * NN * DPAD;

    // hoist loop-invariant weight fragments (W constant over all 20 neighbor slots)
    v16bf bh[BHN];
    if constexpr (HOIST) {
#pragma unroll
        for (int q = 0; q < TPW; ++q) {
            int nt = w + q * 8;
            if (nt < NTILES) {
#pragma unroll
                for (int kc = 0; kc < NCH; ++kc)
                    bh[q * NCH + kc] = load_b_frag(wb, DOUT, kc * 32, nt * 16);
            }
        }
    }

    v8f rmax[TPW];
#pragma unroll
    for (int q = 0; q < TPW; ++q)
#pragma unroll
        for (int v = 0; v < 8; ++v) rmax[q][v] = -INFINITY;

    for (int t = 0; t < KNN; ++t) {
        __syncthreads();
        if (tid < 16) Jn[tid] = idx[((size_t)b * NN + i0 + tid) * KNN + t];
        __syncthreads();
#pragma unroll 4
        for (int e = tid; e < 16 * KPAD; e += 256) {
            int m = e / KPAD, c = e - m * KPAD;
            int gi = i0 + m;
            unsigned short hv;
            if (c < DIN) {
                hv = Xb[(size_t)gi * DPAD + c];
            } else if (c < 2 * DIN) {
                int cc = c - DIN;
                float diff = bf2f(Xb[(size_t)Jn[m] * DPAD + cc]) -
                             bf2f(Xb[(size_t)gi * DPAD + cc]);
                hv = f2bf(diff);
            } else hv = 0;
            E[m * KPAD + c] = hv;
        }
        __syncthreads();

        v8f acc[TPW];
#pragma unroll
        for (int q = 0; q < TPW; ++q) acc[q] = (v8f){};
#pragma unroll
        for (int kc = 0; kc < NCH; ++kc) {
            v16bf a = load_a_frag(E, KPAD, 0, kc * 32);   // ds_load from LDS (shared by tiles)
#pragma unroll
            for (int q = 0; q < TPW; ++q) {
                int nt = w + q * 8;
                if (nt < NTILES) {
                    v16bf bf;
                    if constexpr (HOIST) bf = bh[q * NCH + kc];
                    else                 bf = load_b_frag(wb, DOUT, kc * 32, nt * 16);
                    acc[q] = wmma_bf16(a, bf, acc[q]);
                }
            }
        }
#pragma unroll
        for (int q = 0; q < TPW; ++q)
#pragma unroll
            for (int v = 0; v < 8; ++v) rmax[q][v] = fmaxf(rmax[q][v], acc[q][v]);
    }

#pragma unroll
    for (int q = 0; q < TPW; ++q) {
        int nt = w + q * 8;
        if (nt >= NTILES) continue;
        int n = nt * 16 + (lane & 15);
        float bn = bias[n];
#pragma unroll
        for (int v = 0; v < 8; ++v) {
            int m = i0 + v + ((lane >> 4) << 3);
            xout[((size_t)b * NN + m) * DOUT + n] = rmax[q][v] + bn;
        }
    }
}

// ---------------- K6: big linear (16384x512 @ 512x1024) + fused BN-stats + per-cloud max/min
// grid (LIN/32, B*NN/64), block 256 = 8 waves arranged 4(m) x 2(n).
// Weight tile (512x32 bf16 = 32KB) staged to LDS once via CDNA5 async copy.
__global__ void k_linear(const unsigned short* __restrict__ cat,
                         const unsigned short* __restrict__ wlb,
                         const float* __restrict__ bl,
                         float* __restrict__ colsum, float* __restrict__ colsumsq,
                         unsigned* __restrict__ ymax, unsigned* __restrict__ ymin) {
    __shared__ __align__(128) unsigned short Bl[512 * 32];
    int tid = threadIdx.x;
    int w = tid >> 5, lane = tid & 31;
    int nb = blockIdx.x * 32;
    int m0 = blockIdx.y * 64 + (w >> 1) * 16;

    // async global->LDS stage of the 512x32 weight tile: 2048 x 16B transfers
#pragma unroll 2
    for (int it = tid; it < 2048; it += 256) {
        int k = it >> 2;
        int c = (it & 3) * 8;
        async_copy_b128(wlb + (size_t)k * LIN + nb + c, &Bl[k * 32 + c]);
    }
    async_wait0();
    __syncthreads();

    int n0l = (w & 1) * 16;                        // local column tile within LDS
    v8f acc = {};
#pragma unroll
    for (int k0 = 0; k0 < 512; k0 += 32) {
        __builtin_prefetch(cat + (size_t)m0 * 512 + k0 + 32, 0, 1);
        v16bf a  = load_a_frag(cat, 512, m0, k0);
        v16bf bf = load_b_frag(Bl, 32, k0, n0l);   // ds_load from LDS
        acc = wmma_bf16(a, bf, acc);
    }
    int n = nb + n0l + (lane & 15);
    float bn = bl[n];
    float s = 0.f, ss = 0.f, mx = -INFINITY, mn = INFINITY;
#pragma unroll
    for (int v = 0; v < 8; ++v) {
        float y = acc[v] + bn;
        s += y; ss += y * y;
        mx = fmaxf(mx, y); mn = fminf(mn, y);
    }
    // combine the two half-lane groups (rows m..m+7 vs m+8..m+15)
    s  += __shfl_xor(s, 16, 32);
    ss += __shfl_xor(ss, 16, 32);
    mx  = fmaxf(mx, __shfl_xor(mx, 16, 32));
    mn  = fminf(mn, __shfl_xor(mn, 16, 32));
    if (lane < 16) {
        atomicAdd(&colsum[n], s);
        atomicAdd(&colsumsq[n], ss);
        int bc = m0 >> 10;                          // 16-row tiles never straddle clouds
        atomicMax(&ymax[bc * LIN + n], ford(mx));
        atomicMin(&ymin[bc * LIN + n], ford(mn));
    }
}

// ---------------- K7: BN+leaky+pool (monotone trick) + head MLP (tiny) -----------
__global__ void k_head(const float* __restrict__ colsum, const float* __restrict__ colsumsq,
                       const unsigned* __restrict__ ymax, const unsigned* __restrict__ ymin,
                       const float* __restrict__ gl, const float* __restrict__ betal,
                       const float* __restrict__ Wm1, const float* __restrict__ bm1,
                       const float* __restrict__ gm1, const float* __restrict__ betam1,
                       const float* __restrict__ Wm2, const float* __restrict__ bm2,
                       float* __restrict__ out) {
    extern __shared__ float sm[];
    float* pooled = sm;                 // 16*1024
    float* h      = sm + BB * LIN;      // 16*512
    int tid = threadIdx.x;
    const float inv = 1.f / (float)(BB * NN);
    for (int n = tid; n < LIN; n += 256) {
        float mean = colsum[n] * inv;
        float var  = colsumsq[n] * inv - mean * mean;
        float sc   = gl[n] * rsqrtf(var + 1e-5f);
        float be   = betal[n];
        for (int b = 0; b < BB; ++b) {
            float vmax = funord(ymax[b * LIN + n]);
            float vmin = funord(ymin[b * LIN + n]);
            float val  = (sc >= 0.f) ? vmax : vmin;   // max of monotone-affine map
            float p    = sc * (val - mean) + be;
            pooled[b * LIN + n] = (p > 0.f) ? p : 0.2f * p;
        }
    }
    __syncthreads();
    for (int o = tid; o < BB * HID; o += 256) {
        int b = o >> 9, f = o & 511;
        float a = bm1[f];
        const float* pr = pooled + b * LIN;
        for (int c = 0; c < LIN; ++c) a += pr[c] * Wm1[(size_t)c * HID + f];
        h[o] = a;
    }
    __syncthreads();
    for (int f = tid; f < HID; f += 256) {
        float m = 0.f;
        for (int b = 0; b < BB; ++b) m += h[b * HID + f];
        m *= (1.f / BB);
        float v = 0.f;
        for (int b = 0; b < BB; ++b) { float d = h[b * HID + f] - m; v += d * d; }
        v *= (1.f / BB);
        float sc = gm1[f] * rsqrtf(v + 1e-5f);
        float be = betam1[f];
        for (int b = 0; b < BB; ++b) {
            float z = sc * (h[b * HID + f] - m) + be;
            h[b * HID + f] = (z > 0.f) ? z : 0.2f * z;
        }
    }
    __syncthreads();
    for (int o = tid; o < BB * OUTD; o += 256) {
        int b = o / OUTD, j = o - b * OUTD;
        float a = bm2[j];
        const float* hr = h + b * HID;
        for (int c = 0; c < HID; ++c) a += hr[c] * Wm2[(size_t)c * OUTD + j];
        out[o] = a;
    }
}

// ---------------- host orchestration ----------------
extern "C" void kernel_launch(void* const* d_in, const int* in_sizes, int n_in,
                              void* d_out, int out_size, void* d_ws, size_t ws_size,
                              hipStream_t stream) {
    (void)in_sizes; (void)n_in; (void)out_size; (void)ws_size;
    const float* pos = (const float*)d_in[0];
    // d_in[1] is k (==20, baked in)
    const float* W0 = (const float*)d_in[2];  const float* bW0 = (const float*)d_in[3];
    const float* W1 = (const float*)d_in[4];  const float* bW1 = (const float*)d_in[5];
    const float* W2 = (const float*)d_in[6];  const float* bW2 = (const float*)d_in[7];
    const float* W3 = (const float*)d_in[8];  const float* bW3 = (const float*)d_in[9];
    const float* Wl = (const float*)d_in[10];  const float* bl = (const float*)d_in[11];
    const float* gl = (const float*)d_in[12];  const float* betal = (const float*)d_in[13];
    const float* Wm1 = (const float*)d_in[14]; const float* bm1 = (const float*)d_in[15];
    const float* gm1 = (const float*)d_in[16]; const float* betam1 = (const float*)d_in[17];
    const float* Wm2 = (const float*)d_in[18]; const float* bm2 = (const float*)d_in[19];
    float* out = (float*)d_out;

    // workspace carve-up
    char* ws = (char*)d_ws;
    size_t off = 0;
    auto carve = [&](size_t bytes) { char* p = ws + off; off += (bytes + 255) & ~(size_t)255; return p; };
    unsigned short* xb   = (unsigned short*)carve((size_t)BB * NN * 256 * 2);
    unsigned short* xbT  = (unsigned short*)carve((size_t)BB * NN * 256 * 2);
    float*          sq   = (float*)carve((size_t)BB * NN * 4);
    float*          gram = (float*)carve((size_t)BB * NN * NN * 4);
    int*            idx  = (int*)carve((size_t)BB * NN * KNN * 4);
    float*          xout = (float*)carve((size_t)BB * NN * 256 * 4);
    unsigned short* cat  = (unsigned short*)carve((size_t)BB * NN * 512 * 2);
    unsigned short* wb   = (unsigned short*)carve((size_t)512 * LIN * 2);
    float*          colsum   = (float*)carve(LIN * 4);
    float*          colsumsq = (float*)carve(LIN * 4);
    unsigned*       ymax = (unsigned*)carve((size_t)BB * LIN * 4);
    unsigned*       ymin = (unsigned*)carve((size_t)BB * LIN * 4);

    const int PTS = BB * NN;                        // 16384 points
    k_init<<<PTS / 256, 256, 0, stream>>>(colsum, colsumsq, ymax, ymin);

    dim3 gg(NN / 64, NN / 32, BB);
    dim3 ge(NN / 16, BB);

    // cast the first input (pos): d=3 padded to 32
    k_cast<<<PTS / 256, 256, 0, stream>>>(pos, 3, 32, xb, xbT, sq, nullptr, 0);

    // ---- layer 1: din=3 dpad=32 kpad=32 dout=64 ----
    k_gram<32><<<gg, 256, 0, stream>>>(xb, xbT, gram);
    k_topk<<<PTS / 256, 256, 0, stream>>>(gram, sq, idx);
    k_wcast<<<(32 * 64 + 255) / 256, 256, 0, stream>>>(W0, 6, 32, 64, wb);
    k_edge<3, 32, 32, 64><<<ge, 256, 0, stream>>>(xb, idx, wb, bW0, xout);
    k_cast<<<PTS / 256, 256, 0, stream>>>(xout, 64, 64, xb, xbT, sq, cat, 0);

    // ---- layer 2: din=64 dpad=64 kpad=128 dout=64 ----
    k_gram<64><<<gg, 256, 0, stream>>>(xb, xbT, gram);
    k_topk<<<PTS / 256, 256, 0, stream>>>(gram, sq, idx);
    k_wcast<<<(128 * 64 + 255) / 256, 256, 0, stream>>>(W1, 128, 128, 64, wb);
    k_edge<64, 64, 128, 64><<<ge, 256, 0, stream>>>(xb, idx, wb, bW1, xout);
    k_cast<<<PTS / 256, 256, 0, stream>>>(xout, 64, 64, xb, xbT, sq, cat, 64);

    // ---- layer 3: din=64 dpad=64 kpad=128 dout=128 ----
    k_gram<64><<<gg, 256, 0, stream>>>(xb, xbT, gram);
    k_topk<<<PTS / 256, 256, 0, stream>>>(gram, sq, idx);
    k_wcast<<<(128 * 128 + 255) / 256, 256, 0, stream>>>(W2, 128, 128, 128, wb);
    k_edge<64, 64, 128, 128><<<ge, 256, 0, stream>>>(xb, idx, wb, bW2, xout);
    k_cast<<<PTS / 256, 256, 0, stream>>>(xout, 128, 128, xb, xbT, sq, cat, 128);

    // ---- layer 4: din=128 dpad=128 kpad=256 dout=256 ----
    k_gram<128><<<gg, 256, 0, stream>>>(xb, xbT, gram);
    k_topk<<<PTS / 256, 256, 0, stream>>>(gram, sq, idx);
    k_wcast<<<(256 * 256 + 255) / 256, 256, 0, stream>>>(W3, 256, 256, 256, wb);
    k_edge<128, 128, 256, 256><<<ge, 256, 0, stream>>>(xb, idx, wb, bW3, xout);
    k_cast<<<PTS / 256, 256, 0, stream>>>(xout, 256, 256, xb, xbT, sq, cat, 256);

    // big linear with fused BN stats + per-cloud max/min (async LDS-staged weights)
    k_wcast<<<(512 * LIN + 255) / 256, 256, 0, stream>>>(Wl, 512, 512, LIN, wb);
    dim3 gl6(LIN / 32, PTS / 64);
    k_linear<<<gl6, 256, 0, stream>>>(cat, wb, bl, colsum, colsumsq, ymax, ymin);

    // head
    size_t smem = (size_t)(BB * LIN + BB * HID) * sizeof(float);   // 96 KB
    k_head<<<1, 256, smem, stream>>>(colsum, colsumsq, ymax, ymin,
                                     gl, betal, Wm1, bm1, gm1, betam1, Wm2, bm2, out);
}
